// MaterialsGNN_32074815766570
// MI455X (gfx1250) — matrix operand order repaired and checked by reading, not compiled, not run on hardware
//
#include <hip/hip_runtime.h>
#include <hip/hip_bf16.h>

#define TPB 256
#define CEILDIV(a,b) (((a)+(b)-1)/(b))

typedef __attribute__((ext_vector_type(16))) _Float16 v16h;
typedef __attribute__((ext_vector_type(8)))  _Float16 v8h;
typedef __attribute__((ext_vector_type(8)))  float    v8f;

// ---------- generic fill ----------
__global__ void k_fill(float* p, float v, long n) {
  long i = (long)blockIdx.x * blockDim.x + threadIdx.x;
  if (i < n) p[i] = v;
}

// ---------- node embedding: h = relu(x @ node_W + node_b) ----------
__global__ void k_node_embed(const float* __restrict__ x, const float* __restrict__ W,
                             const float* __restrict__ b, float* __restrict__ h, int N) {
  long i = (long)blockIdx.x * blockDim.x + threadIdx.x;
  if (i >= (long)N * 64) return;
  int n = i >> 6, d = i & 63;
  float s = b[d];
  #pragma unroll
  for (int j = 0; j < 3; ++j) s += x[n * 3 + j] * W[j * 64 + d];
  h[i] = fmaxf(s, 0.f);
}

// ---------- degree over dst ----------
__global__ void k_deg(const int* __restrict__ dst, float* __restrict__ deg, int E) {
  int e = blockIdx.x * blockDim.x + threadIdx.x;
  if (e < E) atomicAdd(&deg[dst[e]], 1.0f);
}

// ---------- ea_self accumulation: scatter relu(edge_attr@eemb_W+b) by dst ----------
__global__ void k_eaself_scatter(const float* __restrict__ eattr, const float* __restrict__ W,
                                 const float* __restrict__ b, const int* __restrict__ dst,
                                 float* __restrict__ ea_self, int E) {
  long i = (long)blockIdx.x * blockDim.x + threadIdx.x;
  if (i >= (long)E * 64) return;
  int e = i >> 6, d = i & 63;
  float s = b[d];
  #pragma unroll
  for (int j = 0; j < 4; ++j) s += eattr[e * 4 + j] * W[j * 64 + d];
  s = fmaxf(s, 0.f);
  atomicAdd(&ea_self[(long)dst[e] * 64 + d], s);
}

__global__ void k_eaself_div(float* __restrict__ ea_self, const float* __restrict__ deg, int N) {
  long i = (long)blockIdx.x * blockDim.x + threadIdx.x;
  if (i >= (long)N * 64) return;
  ea_self[i] /= fmaxf(deg[i >> 6], 1.0f);
}

// ---------- pre-contract attention weights: aW[d,h] = sum_c W[d, h*64+c]*a[h,c] ----------
__global__ void k_reduce_attw(const float* __restrict__ linW, const float* __restrict__ asrc,
                              const float* __restrict__ adst, const float* __restrict__ lineW,
                              const float* __restrict__ ae, float* __restrict__ asW,
                              float* __restrict__ adW, float* __restrict__ aeW) {
  int i = blockIdx.x * blockDim.x + threadIdx.x;
  if (i >= 256) return;
  int d = i >> 2, hh = i & 3;
  float s1 = 0.f, s2 = 0.f, s3 = 0.f;
  for (int c = 0; c < 64; ++c) {
    s1 += linW[d * 256 + hh * 64 + c] * asrc[hh * 64 + c];
    s2 += linW[d * 256 + hh * 64 + c] * adst[hh * 64 + c];
    s3 += lineW[d * 256 + hh * 64 + c] * ae[hh * 64 + c];
  }
  asW[d * 4 + hh] = s1; adW[d * 4 + hh] = s2; aeW[d * 4 + hh] = s3;
}

// ---------- f32 -> f16 conversion (row-major, for A) ----------
__global__ void k_cvt_f16(const float* __restrict__ src, _Float16* __restrict__ dst, long n) {
  long i = (long)blockIdx.x * blockDim.x + threadIdx.x;
  if (i < n) dst[i] = (_Float16)src[i];
}

// ---------- pack W[64,256] (f32) into WMMA-B lane layout (f16) ----------
// pack[(kbi*16 + nt)*512 + lane*16 + j] = W[(kbi*32 + (lane<16?0:8) + (j<8?j:j+8))*256 + nt*16 + (lane&15)]
__global__ void k_pack_B(const float* __restrict__ W, _Float16* __restrict__ pack) {
  int idx = blockIdx.x * blockDim.x + threadIdx.x;
  if (idx >= 16384) return;
  int kbi = idx >> 13;
  int r   = idx & 8191;
  int nt  = r >> 9;
  int r2  = r & 511;
  int l   = r2 >> 4;
  int j   = r2 & 15;
  int col = nt * 16 + (l & 15);
  int k   = kbi * 32 + ((l < 16) ? 0 : 8) + ((j < 8) ? j : j + 8);
  pack[idx] = (_Float16)W[k * 256 + col];
}

// ---------- WMMA GEMM: xp[N,256] = h16[N,64] @ W (B pre-packed) ----------
// block = one 16-row M tile, 16 waves = 16 column tiles; B staged in LDS.
__global__ void __launch_bounds__(512)
k_gemm_xp_wmma(const _Float16* __restrict__ A16, const _Float16* __restrict__ Bpack,
               float* __restrict__ C, int N) {
  __shared__ __align__(32) _Float16 Bs[16384];  // 32 KB: [kbi][nt][lane][j]
  // cooperative staging: 2048 v8h chunks, 512 threads -> 4 each
  {
    const v8h* gsrc = (const v8h*)Bpack;
    v8h* ldst = (v8h*)Bs;
    #pragma unroll
    for (int t = 0; t < 4; ++t) ldst[threadIdx.x + 512 * t] = gsrc[threadIdx.x + 512 * t];
  }
  __syncthreads();

  int lane = threadIdx.x & 31;
  int nt   = threadIdx.x >> 5;          // 0..15 column tile
  int row0 = blockIdx.x * 16;           // M tile

  v8f acc = {};
  #pragma unroll
  for (int kbi = 0; kbi < 2; ++kbi) {
    // A: lane l(0..15) = row m=l holds K {0..7,16..23}; lane l+16 = row m=l, K {8..15,24..31}
    int row = row0 + (lane & 15);
    row = row < N ? row : N - 1;        // clamp (no predication -> no exec branches)
    const _Float16* arow = A16 + (long)row * 64 + kbi * 32 + ((lane < 16) ? 0 : 8);
    v8h alo = *(const v8h*)(arow);
    v8h ahi = *(const v8h*)(arow + 16);
    v16h a = __builtin_shufflevector(alo, ahi, 0, 1, 2, 3, 4, 5, 6, 7,
                                     8, 9, 10, 11, 12, 13, 14, 15);
    v16h b = *(const v16h*)&Bs[(kbi * 16 + nt) * 512 + lane * 16];
    acc = __builtin_amdgcn_wmma_f32_16x16x32_f16(false, a, false, b, (short)0, acc,
                                                 false, false);
  }
  // D: lane l: col = l%16, rows r + (l<16 ? 0 : 8)
  int col = nt * 16 + (lane & 15);
  int rbase = row0 + ((lane < 16) ? 0 : 8);
  if (row0 + 16 <= N) {
    // full tile (uniform branch): unguarded coalesced stores
    float* crow = C + (long)rbase * 256 + col;
    #pragma unroll
    for (int r = 0; r < 8; ++r) crow[(long)r * 256] = acc[r];
  } else {
    #pragma unroll
    for (int r = 0; r < 8; ++r) {
      int row = rbase + r;
      if (row < N) C[(long)row * 256 + col] = acc[r];
    }
  }
}

// ---------- per-node attention logits a_s, a_d ----------
__global__ void k_asad(const float* __restrict__ h, const float* __restrict__ asW,
                       const float* __restrict__ adW, float* __restrict__ a_s,
                       float* __restrict__ a_d, int N) {
  long i = (long)blockIdx.x * blockDim.x + threadIdx.x;
  if (i >= (long)N * 4) return;
  int n = i >> 2, hh = i & 3;
  float s1 = 0.f, s2 = 0.f;
  for (int d = 0; d < 64; ++d) {
    float hv = h[(long)n * 64 + d];
    s1 += hv * asW[d * 4 + hh];
    s2 += hv * adW[d * 4 + hh];
  }
  a_s[i] = s1; a_d[i] = s2;
}

__device__ inline void atomicMaxF(float* addr, float val) {
  unsigned int* u = (unsigned int*)addr;
  unsigned int old = *u;
  while (__uint_as_float(old) < val) {
    unsigned int prev = atomicCAS(u, old, __float_as_uint(val));
    if (prev == old) break;
    old = prev;
  }
}

// ---------- alpha = leaky_relu(a_s[src]+a_d[dst]+a_ed) ; segment max into m ----------
__global__ void k_alpha(const int* __restrict__ src, const int* __restrict__ dst,
                        const float* __restrict__ eattr, const float* __restrict__ eW,
                        const float* __restrict__ eb, const float* __restrict__ ea_self,
                        const float* __restrict__ aeW, const float* __restrict__ a_s,
                        const float* __restrict__ a_d, float* __restrict__ alpha,
                        float* __restrict__ m, int E, int N) {
  int e = blockIdx.x * blockDim.x + threadIdx.x;
  if (e >= E + N) return;
  int s, t;
  float acc0 = 0.f, acc1 = 0.f, acc2 = 0.f, acc3 = 0.f;
  if (e < E) {
    s = src[e]; t = dst[e];
    float e0 = eattr[e * 4 + 0], e1 = eattr[e * 4 + 1];
    float e2 = eattr[e * 4 + 2], e3 = eattr[e * 4 + 3];
    for (int d = 0; d < 64; ++d) {
      float v = eb[d] + e0 * eW[d] + e1 * eW[64 + d] + e2 * eW[128 + d] + e3 * eW[192 + d];
      v = fmaxf(v, 0.f);
      acc0 += v * aeW[d * 4 + 0]; acc1 += v * aeW[d * 4 + 1];
      acc2 += v * aeW[d * 4 + 2]; acc3 += v * aeW[d * 4 + 3];
    }
  } else {
    int n = e - E; s = n; t = n;
    const float* er = ea_self + (long)n * 64;
    for (int d = 0; d < 64; ++d) {
      float v = er[d];
      acc0 += v * aeW[d * 4 + 0]; acc1 += v * aeW[d * 4 + 1];
      acc2 += v * aeW[d * 4 + 2]; acc3 += v * aeW[d * 4 + 3];
    }
  }
  float accs[4] = {acc0, acc1, acc2, acc3};
  #pragma unroll
  for (int hh = 0; hh < 4; ++hh) {
    float al = a_s[s * 4 + hh] + a_d[t * 4 + hh] + accs[hh];
    al = al > 0.f ? al : 0.2f * al;
    alpha[(long)e * 4 + hh] = al;
    atomicMaxF(&m[t * 4 + hh], al);
  }
}

// ---------- ex = exp(alpha - m[dst]); den += ex ----------
__global__ void k_ex(const int* __restrict__ dst, float* __restrict__ alpha,
                     const float* __restrict__ m, float* __restrict__ den, int E, int N) {
  long i = (long)blockIdx.x * blockDim.x + threadIdx.x;
  if (i >= (long)(E + N) * 4) return;
  int e = i >> 2, hh = i & 3;
  int t = (e < E) ? dst[e] : (e - E);
  float ex = __expf(alpha[i] - m[t * 4 + hh]);
  alpha[i] = ex;
  atomicAdd(&den[t * 4 + hh], ex);
}

// ---------- att = ex / den[dst] ----------
__global__ void k_att(const int* __restrict__ dst, float* __restrict__ alpha,
                      const float* __restrict__ den, int E, int N) {
  long i = (long)blockIdx.x * blockDim.x + threadIdx.x;
  if (i >= (long)(E + N) * 4) return;
  int e = i >> 2, hh = i & 3;
  int t = (e < E) ? dst[e] : (e - E);
  alpha[i] = alpha[i] / den[t * 4 + hh];
}

// ---------- message scatter: out[dst] += xp[src] * att ----------
__global__ void k_scatter(const int* __restrict__ src, const int* __restrict__ dst,
                          const float* __restrict__ xp, const float* __restrict__ alpha,
                          float* __restrict__ outacc, int E, int N) {
  long i = (long)blockIdx.x * blockDim.x + threadIdx.x;
  if (i >= (long)(E + N) * 64) return;
  int e = (int)(i >> 6), c = (int)(i & 63);
  int s = (e < E) ? src[e] : (e - E);
  int t = (e < E) ? dst[e] : (e - E);
  const float* xr = xp + (long)s * 256 + c;
  float* orow = outacc + (long)t * 256 + c;
  __builtin_prefetch(xr, 0, 3);
  #pragma unroll
  for (int hh = 0; hh < 4; ++hh)
    atomicAdd(orow + hh * 64, xr[hh * 64] * alpha[(long)e * 4 + hh]);
}

// ---------- head mean + bias + relu -> h ----------
__global__ void k_headmean(const float* __restrict__ outacc, const float* __restrict__ cb,
                           float* __restrict__ h, int N) {
  long i = (long)blockIdx.x * blockDim.x + threadIdx.x;
  if (i >= (long)N * 64) return;
  int n = (int)(i >> 6), c = (int)(i & 63);
  const float* o = outacc + (long)n * 256 + c;
  float v = 0.25f * (o[0] + o[64] + o[128] + o[192]) + cb[c];
  h[i] = fmaxf(v, 0.f);
}

// ---------- pooling ----------
__global__ void k_cnt(const int* __restrict__ batch, float* __restrict__ cnt, int N) {
  int n = blockIdx.x * blockDim.x + threadIdx.x;
  if (n < N) atomicAdd(&cnt[batch[n]], 1.0f);
}
__global__ void k_pool(const int* __restrict__ batch, const float* __restrict__ h,
                       float* __restrict__ pooled, int N) {
  long i = (long)blockIdx.x * blockDim.x + threadIdx.x;
  if (i >= (long)N * 64) return;
  int n = (int)(i >> 6), d = (int)(i & 63);
  atomicAdd(&pooled[(long)batch[n] * 64 + d], h[i]);
}
__global__ void k_pooldiv(float* __restrict__ pooled, const float* __restrict__ cnt, int B) {
  int i = blockIdx.x * blockDim.x + threadIdx.x;
  if (i >= B * 64) return;
  pooled[i] /= fmaxf(cnt[i >> 6], 1.0f);
}
__global__ void k_ug(const float* __restrict__ u, const float* __restrict__ gW,
                     const float* __restrict__ gb, float* __restrict__ ug, int B) {
  int i = blockIdx.x * blockDim.x + threadIdx.x;
  if (i >= B * 64) return;
  int b = i >> 6, d = i & 63;
  float s = gb[d];
  for (int j = 0; j < 10; ++j) s += u[b * 10 + j] * gW[j * 64 + d];
  ug[i] = fmaxf(s, 0.f);
}
__global__ void k_z1(const float* __restrict__ pooled, const float* __restrict__ ug,
                     const float* __restrict__ f1W, const float* __restrict__ f1b,
                     float* __restrict__ z1, int B) {
  int i = blockIdx.x * blockDim.x + threadIdx.x;
  if (i >= B * 64) return;
  int b = i >> 6, d = i & 63;
  float s = f1b[d];
  for (int k = 0; k < 64; ++k) s += pooled[b * 64 + k] * f1W[k * 64 + d];
  for (int k = 0; k < 64; ++k) s += ug[b * 64 + k] * f1W[(64 + k) * 64 + d];
  z1[i] = fmaxf(s, 0.f);
}
__global__ void k_out(const float* __restrict__ z1, const float* __restrict__ f2W,
                      const float* __restrict__ f2b, float* __restrict__ out, int B) {
  int i = blockIdx.x * blockDim.x + threadIdx.x;
  if (i >= B * 2) return;
  int b = i >> 1, o = i & 1;
  float s = f2b[o];
  for (int k = 0; k < 64; ++k) s += z1[b * 64 + k] * f2W[k * 2 + o];
  out[i] = s;
}

extern "C" void kernel_launch(void* const* d_in, const int* in_sizes, int n_in,
                              void* d_out, int out_size, void* d_ws, size_t ws_size,
                              hipStream_t stream) {
  const float* x      = (const float*)d_in[0];
  const int*   ei     = (const int*)  d_in[1];
  const float* eattr  = (const float*)d_in[2];
  const float* u      = (const float*)d_in[3];
  const int*   batch  = (const int*)  d_in[4];
  const float* nodeW  = (const float*)d_in[5];
  const float* nodeb  = (const float*)d_in[6];
  const float* eembW  = (const float*)d_in[7];
  const float* eembb  = (const float*)d_in[8];
  const float* linW   = (const float*)d_in[9];
  const float* attsrc = (const float*)d_in[10];
  const float* attdst = (const float*)d_in[11];
  const float* lineW  = (const float*)d_in[12];
  const float* atte   = (const float*)d_in[13];
  const float* convb  = (const float*)d_in[14];
  const float* gW     = (const float*)d_in[15];
  const float* gb     = (const float*)d_in[16];
  const float* f1W    = (const float*)d_in[17];
  const float* f1b    = (const float*)d_in[18];
  const float* f2W    = (const float*)d_in[19];
  const float* f2b    = (const float*)d_in[20];

  const int N = in_sizes[4];
  const int E = in_sizes[1] / 2;
  const int B = in_sizes[3] / 10;
  const int E2 = E + N;
  const int* srcI = ei;
  const int* dstI = ei + E;

  // workspace carving (256B aligned)
  char* base = (char*)d_ws;
  size_t off = 0;
  auto carve = [&](size_t bytes) -> char* {
    off = (off + 255) & ~(size_t)255;
    char* p = base + off;
    off += bytes;
    return p;
  };
  float*     h       = (float*)    carve((size_t)N * 64 * 4);
  _Float16*  h16     = (_Float16*) carve((size_t)N * 64 * 2);
  _Float16*  Wpack   = (_Float16*) carve((size_t)16384 * 2);
  float*     xp      = (float*)    carve((size_t)N * 256 * 4);
  float*     outacc  = (float*)    carve((size_t)N * 256 * 4);
  float*     ea_self = (float*)    carve((size_t)N * 64 * 4);
  float*     deg     = (float*)    carve((size_t)N * 4);
  float*     a_s     = (float*)    carve((size_t)N * 4 * 4);
  float*     a_d     = (float*)    carve((size_t)N * 4 * 4);
  float*     mbuf    = (float*)    carve((size_t)N * 4 * 4);
  float*     den     = (float*)    carve((size_t)N * 4 * 4);
  float*     alpha   = (float*)    carve((size_t)E2 * 4 * 4);
  float*     asW     = (float*)    carve(256 * 4);
  float*     adW     = (float*)    carve(256 * 4);
  float*     aeW     = (float*)    carve(256 * 4);
  float*     pooled  = (float*)    carve((size_t)B * 64 * 4);
  float*     cnt     = (float*)    carve((size_t)B * 4);
  float*     ug      = (float*)    carve((size_t)B * 64 * 4);
  float*     z1      = (float*)    carve((size_t)B * 64 * 4);
  (void)ws_size; (void)n_in;

  // node + edge embeddings (edge self-loop attr = mean of incoming)
  k_node_embed<<<CEILDIV((long)N * 64, TPB), TPB, 0, stream>>>(x, nodeW, nodeb, h, N);
  k_fill<<<CEILDIV((long)N, TPB), TPB, 0, stream>>>(deg, 0.f, N);
  k_fill<<<CEILDIV((long)N * 64, TPB), TPB, 0, stream>>>(ea_self, 0.f, (long)N * 64);
  k_deg<<<CEILDIV(E, TPB), TPB, 0, stream>>>(dstI, deg, E);
  k_eaself_scatter<<<CEILDIV((long)E * 64, TPB), TPB, 0, stream>>>(eattr, eembW, eembb, dstI,
                                                                   ea_self, E);
  k_eaself_div<<<CEILDIV((long)N * 64, TPB), TPB, 0, stream>>>(ea_self, deg, N);

  for (int layer = 0; layer < 3; ++layer) {
    const float* lW  = linW  + (size_t)layer * 64 * 256;
    const float* leW = lineW + (size_t)layer * 64 * 256;
    const float* asc = attsrc + (size_t)layer * 256;
    const float* adc = attdst + (size_t)layer * 256;
    const float* aec = atte  + (size_t)layer * 256;
    const float* cb  = convb + (size_t)layer * 64;

    k_reduce_attw<<<1, 256, 0, stream>>>(lW, asc, adc, leW, aec, asW, adW, aeW);
    k_cvt_f16<<<CEILDIV((long)N * 64, TPB), TPB, 0, stream>>>(h, h16, (long)N * 64);
    k_pack_B<<<CEILDIV(16384, TPB), TPB, 0, stream>>>(lW, Wpack);
    k_gemm_xp_wmma<<<CEILDIV(N, 16), 512, 0, stream>>>(h16, Wpack, xp, N);
    k_asad<<<CEILDIV((long)N * 4, TPB), TPB, 0, stream>>>(h, asW, adW, a_s, a_d, N);

    k_fill<<<CEILDIV((long)N * 4, TPB), TPB, 0, stream>>>(mbuf, -3.0e38f, (long)N * 4);
    k_fill<<<CEILDIV((long)N * 4, TPB), TPB, 0, stream>>>(den, 0.f, (long)N * 4);
    k_fill<<<CEILDIV((long)N * 256, TPB), TPB, 0, stream>>>(outacc, 0.f, (long)N * 256);

    k_alpha<<<CEILDIV(E2, TPB), TPB, 0, stream>>>(srcI, dstI, eattr, eembW, eembb, ea_self,
                                                  aeW, a_s, a_d, alpha, mbuf, E, N);
    k_ex<<<CEILDIV((long)E2 * 4, TPB), TPB, 0, stream>>>(dstI, alpha, mbuf, den, E, N);
    k_att<<<CEILDIV((long)E2 * 4, TPB), TPB, 0, stream>>>(dstI, alpha, den, E, N);
    k_scatter<<<CEILDIV((long)E2 * 64, TPB), TPB, 0, stream>>>(srcI, dstI, xp, alpha,
                                                               outacc, E, N);
    k_headmean<<<CEILDIV((long)N * 64, TPB), TPB, 0, stream>>>(outacc, cb, h, N);
  }

  // global mean pool + MLP head
  k_fill<<<CEILDIV((long)B * 64, TPB), TPB, 0, stream>>>(pooled, 0.f, (long)B * 64);
  k_fill<<<1, TPB, 0, stream>>>(cnt, 0.f, B);
  k_cnt<<<CEILDIV(N, TPB), TPB, 0, stream>>>(batch, cnt, N);
  k_pool<<<CEILDIV((long)N * 64, TPB), TPB, 0, stream>>>(batch, h, pooled, N);
  k_pooldiv<<<CEILDIV(B * 64, TPB), TPB, 0, stream>>>(pooled, cnt, B);
  k_ug<<<CEILDIV(B * 64, TPB), TPB, 0, stream>>>(u, gW, gb, ug, B);
  k_z1<<<CEILDIV(B * 64, TPB), TPB, 0, stream>>>(pooled, ug, f1W, f1b, z1, B);
  k_out<<<CEILDIV(B * 2, TPB), TPB, 0, stream>>>(z1, f2W, f2b, (float*)d_out, B);
}
